// MoeExpertChoice_40123584479378
// MI455X (gfx1250) — compile-verified, matrix operand order checked by reading, not run on hardware
//
#include <hip/hip_runtime.h>
#include <hip/hip_bf16.h>
#include <math.h>

// Problem constants (match reference)
constexpr int Bc  = 8;
constexpr int Sc  = 4096;
constexpr int Dc  = 128;   // d_in
constexpr int Hc  = 2048;  // hidden
constexpr int Ec  = 64;    // experts
constexpr int Kc  = 64;    // topk tokens per expert
constexpr int DOc = 128;   // d_out
constexpr int HT  = 64;    // hidden chunk processed per fused-loop iteration

typedef __attribute__((ext_vector_type(16))) __bf16 v16bf;
typedef __attribute__((ext_vector_type(8)))  __bf16 v8bf;
typedef __attribute__((ext_vector_type(8)))  float  v8f;

// ---------------------------------------------------------------------------
// WMMA bf16 fragment loader.
// LDS tile layout is row-major [row][k] with `stride` elements per row
// (stride must be a multiple of 8 so each half-fragment is a 16B-aligned b128).
// Per ISA 7.12.2 (16-bit A 16x32): lane<16 -> row=lane, K elems {0..7,16..23};
// lane>=16 -> row=lane-16, K elems {8..15,24..31}. B (32x16) mirrors with
// row = output column, so storing B tiles K-major lets one loader serve both.
// ---------------------------------------------------------------------------
__device__ __forceinline__ v16bf load_frag(const __bf16* lds, int r0, int k0, int stride) {
    int lane = threadIdx.x & 31;
    int r    = r0 + (lane & 15);
    int kb0  = k0 + ((lane >> 4) << 3);      // k0 + half*8
    const __bf16* p = lds + r * stride + kb0;
    v8bf lo = *(const v8bf*)(p);             // K = kb0 .. kb0+7
    v8bf hi = *(const v8bf*)(p + 16);        // K = kb0+16 .. kb0+23
    return __builtin_shufflevector(lo, hi, 0,1,2,3,4,5,6,7,8,9,10,11,12,13,14,15);
}

__device__ __forceinline__ v8f wmma_bf16(v16bf a, v16bf b, v8f c) {
    return __builtin_amdgcn_wmma_f32_16x16x32_bf16(false, a, false, b, (short)0, c, false, false);
}

// ---------------------------------------------------------------------------
// Kernel 0: zero the output (harness poisons d_out; we scatter-add into it)
// ---------------------------------------------------------------------------
__global__ void k_zero(float* __restrict__ p, int n) {
    int i = blockIdx.x * blockDim.x + threadIdx.x;
    for (; i < n; i += gridDim.x * blockDim.x) p[i] = 0.0f;
}

// ---------------------------------------------------------------------------
// Kernel 1: gate logits via WMMA, stored TRANSPOSED as logitsT[b][e][s]
// so that softmax-over-tokens and top-k are contiguous.
// One block = 64 tokens of one batch x all 64 experts.
// ---------------------------------------------------------------------------
__global__ __launch_bounds__(256) void k_gate(const float* __restrict__ x,
                                              const float* __restrict__ gw,
                                              const float* __restrict__ gb,
                                              float* __restrict__ logitsT) {
    constexpr int TS = 64;
    int blk = blockIdx.x;                  // 0 .. B*S/TS-1
    int b   = blk >> 6;                    // S/TS = 64 blocks per batch
    int t0  = (blk & 63) * TS;

    __shared__ __align__(16) __bf16 aT[TS][Dc + 8];   // tokens x d   (row-major K)
    __shared__ __align__(16) __bf16 bT[Ec][Dc + 8];   // experts x d  (K-major B tile)
    __shared__ float scores[Ec][TS + 4];              // transpose buffer
    __shared__ float gbS[Ec];

    int tid = threadIdx.x;
    const float* xb = x + ((size_t)b * Sc + t0) * Dc;
    for (int i = tid; i < TS * Dc; i += 256) {
        int r = i >> 7, c = i & 127;
        aT[r][c] = (__bf16)xb[r * Dc + c];
    }
    for (int i = tid; i < Dc * Ec; i += 256) {         // gw is [D][E]
        int e = i & 63, d = i >> 6;
        bT[e][d] = (__bf16)gw[(size_t)d * Ec + e];
    }
    if (tid < Ec) gbS[tid] = gb[tid];
    __syncthreads();

    int wave = tid >> 5;
    int m0   = (wave & 3) * 16;                        // token strip
    int ntB  = (wave >> 2) * 2;                        // 2 expert tiles per wave
    v8f acc0 = {}, acc1 = {};
    #pragma unroll
    for (int kk = 0; kk < Dc; kk += 32) {
        v16bf a  = load_frag(&aT[0][0], m0, kk, Dc + 8);
        v16bf b0 = load_frag(&bT[0][0], ntB * 16,       kk, Dc + 8);
        v16bf b1 = load_frag(&bT[0][0], (ntB + 1) * 16, kk, Dc + 8);
        acc0 = wmma_bf16(a, b0, acc0);
        acc1 = wmma_bf16(a, b1, acc1);
    }
    int lane = tid & 31;
    int nL   = lane & 15;
    int mAdd = (lane >> 4) * 8;
    #pragma unroll
    for (int i = 0; i < 8; ++i) {
        int m = m0 + mAdd + i;
        scores[ntB * 16 + nL][m]       = acc0[i];
        scores[(ntB + 1) * 16 + nL][m] = acc1[i];
    }
    __syncthreads();
    // coalesced write: consecutive tid -> consecutive s
    for (int i = tid; i < Ec * TS; i += 256) {
        int e = i >> 6, s = i & 63;
        logitsT[((size_t)b * Ec + e) * Sc + t0 + s] = scores[e][s] + gbS[e];
    }
}

// ---------------------------------------------------------------------------
// Kernel 2: softmax over the TOKEN axis. One block per (b,e); row of S=4096
// contiguous floats. Overwrites logits with probs in place.
// ---------------------------------------------------------------------------
__global__ __launch_bounds__(256) void k_softmax(float* __restrict__ logitsT) {
    float* row = logitsT + (size_t)blockIdx.x * Sc;
    __shared__ float red[256];
    int tid = threadIdx.x;

    float mx = -INFINITY;
    #pragma unroll
    for (int j = 0; j < Sc / 256; ++j) mx = fmaxf(mx, row[tid + j * 256]);
    red[tid] = mx; __syncthreads();
    for (int o = 128; o > 0; o >>= 1) {
        if (tid < o) red[tid] = fmaxf(red[tid], red[tid + o]);
        __syncthreads();
    }
    mx = red[0]; __syncthreads();

    float loc[Sc / 256];
    float sum = 0.0f;
    #pragma unroll
    for (int j = 0; j < Sc / 256; ++j) {
        loc[j] = __expf(row[tid + j * 256] - mx);
        sum += loc[j];
    }
    red[tid] = sum; __syncthreads();
    for (int o = 128; o > 0; o >>= 1) {
        if (tid < o) red[tid] += red[tid + o];
        __syncthreads();
    }
    float inv = 1.0f / red[0];
    #pragma unroll
    for (int j = 0; j < Sc / 256; ++j) row[tid + j * 256] = loc[j] * inv;
}

// ---------------------------------------------------------------------------
// Kernel 3: top-K=64 tokens per (b,e). Block-local iterative argmax over the
// LDS-resident 4096 probs (probs >= 0, so -1 is a safe tombstone).
// Ties resolved toward lower index (matches lax.top_k's first-occurrence).
// ---------------------------------------------------------------------------
__global__ __launch_bounds__(256) void k_topk(const float* __restrict__ probsT,
                                              float* __restrict__ topv,
                                              int*   __restrict__ topi) {
    int be = blockIdx.x;
    const float* row = probsT + (size_t)be * Sc;
    __shared__ float vals[Sc];
    __shared__ float rv[256];
    __shared__ int   ri[256];
    int tid = threadIdx.x;
    for (int i = tid; i < Sc; i += 256) vals[i] = row[i];
    __syncthreads();

    for (int k = 0; k < Kc; ++k) {
        float bv = -1.0f; int bi = tid * 16;
        #pragma unroll
        for (int j = 0; j < 16; ++j) {          // contiguous chunk -> stable tie order
            int idx = tid * 16 + j;
            float v = vals[idx];
            if (v > bv) { bv = v; bi = idx; }
        }
        rv[tid] = bv; ri[tid] = bi; __syncthreads();
        for (int o = 128; o > 0; o >>= 1) {
            if (tid < o) {
                bool take = (rv[tid + o] > rv[tid]) ||
                            (rv[tid + o] == rv[tid] && ri[tid + o] < ri[tid]);
                if (take) { rv[tid] = rv[tid + o]; ri[tid] = ri[tid + o]; }
            }
            __syncthreads();
        }
        if (tid == 0) {
            topv[(size_t)be * Kc + k] = rv[0];
            topi[(size_t)be * Kc + k] = ri[0];
            vals[ri[0]] = -1.0f;
        }
        __syncthreads();
    }
}

// ---------------------------------------------------------------------------
// Kernel 4: fused per-expert MLP, software-pipelined.
// Block = (expert e, 128-row tile); with ~1 block per WGP there is no
// cross-block latency hiding, so chunk hc+1's weights are prefetched into
// registers while chunk hc's WMMAs run; the fp32->bf16 convert + LDS commit
// happens behind the stage-2 barrier. The 134 MB `h` never touches HBM.
// ---------------------------------------------------------------------------
__global__ __launch_bounds__(256) void k_expert(const float* __restrict__ x,
                                                const float* __restrict__ w1,
                                                const float* __restrict__ w2,
                                                const float* __restrict__ topv,
                                                const int*   __restrict__ topi,
                                                float* __restrict__ out) {
    int e  = blockIdx.x >> 2;
    int r0 = (blockIdx.x & 3) * 128;

    __shared__ __align__(16) __bf16 aT [128][Dc + 8];   // tokens x D    (A, row-major K)
    __shared__ __align__(16) __bf16 w1T[HT ][Dc + 8];   // hcols x D     (B1, K-major)
    __shared__ __align__(16) __bf16 hT [128][HT + 8];   // tokens x HT   (A2, row-major K)
    __shared__ __align__(16) __bf16 w2T[DOc][HT + 8];   // ocols x HT    (B2, K-major)
    __shared__ float bias1[HT];
    __shared__ float bias2[DOc];
    __shared__ int   rowTok[128];
    __shared__ int   rowB[128];
    __shared__ float rowGate[128];

    int tid = threadIdx.x;

    // Row metadata: row r of expert e corresponds to (b = r/K, k = r%K),
    // token s = topi[b,e,k], gate = topv[b,e,k].
    if (tid < 128) {
        int rr = r0 + tid;
        int b = rr >> 6, k = rr & 63;
        size_t off = ((size_t)b * Ec + e) * Kc + k;
        rowTok[tid]  = topi[off];
        rowGate[tid] = topv[off];
        rowB[tid]    = b;
    }
    __syncthreads();

    // Gather selected token rows of x -> bf16 A tile
    for (int i = tid; i < 128 * Dc; i += 256) {
        int r = i >> 7, c = i & 127;
        aT[r][c] = (__bf16)x[(((size_t)rowB[r] * Sc + rowTok[r]) * Dc) + c];
    }
    const size_t w1e = (size_t)e * (Dc + 1) * Hc;
    const size_t w2e = (size_t)e * (Hc + 1) * DOc;
    // bias2 = appended-ones row of weight2 (loop-invariant)
    if (tid < DOc) bias2[tid] = w2[w2e + (size_t)Hc * DOc + tid];

    int wave = tid >> 5;
    int m0   = wave * 16;
    int lane = tid & 31;
    int nL   = lane & 15;
    int mAdd = (lane >> 4) * 8;

    v8f oacc[8] = {};

    // --- register prefetch buffers (double-buffer the weight stream) -------
    float pw1[32], pw2[32], pb1 = 0.0f;
    auto prefetch = [&](int hc) {
        #pragma unroll
        for (int j = 0; j < 32; ++j) {                 // w1[e][d][hc+n]
            int i = tid + 256 * j;
            int n = i & 63, d = i >> 6;
            pw1[j] = w1[w1e + (size_t)d * Hc + hc + n];
        }
        #pragma unroll
        for (int j = 0; j < 32; ++j) {                 // w2[e][hc+k][n]
            int i = tid + 256 * j;
            int n = i & 127, k = i >> 7;
            pw2[j] = w2[w2e + (size_t)(hc + k) * DOc + n];
        }
        if (tid < HT) pb1 = w1[w1e + (size_t)Dc * Hc + hc + tid];
    };
    auto commit = [&]() {
        #pragma unroll
        for (int j = 0; j < 32; ++j) {
            int i = tid + 256 * j;
            w1T[i & 63][i >> 6] = (__bf16)pw1[j];      // K-major B1 tile
        }
        #pragma unroll
        for (int j = 0; j < 32; ++j) {
            int i = tid + 256 * j;
            w2T[i & 127][i >> 7] = (__bf16)pw2[j];     // K-major B2 tile
        }
        if (tid < HT) bias1[tid] = pb1;
    };

    prefetch(0);
    commit();
    __syncthreads();

    for (int hc = 0; hc < Hc; hc += HT) {
        // Issue next chunk's global loads now; they complete during the WMMAs.
        if (hc + HT < Hc) prefetch(hc + HT);

        // Stage 1: 16-row h strip = A(16x128) @ W1c(128x64), bias + exact GELU
        #pragma unroll
        for (int nt = 0; nt < 4; ++nt) {
            v8f acc = {};
            #pragma unroll
            for (int kk = 0; kk < Dc; kk += 32) {
                v16bf a  = load_frag(&aT[0][0],  m0,      kk, Dc + 8);
                v16bf bb = load_frag(&w1T[0][0], nt * 16, kk, Dc + 8);
                acc = wmma_bf16(a, bb, acc);
            }
            int nn = nt * 16 + nL;
            float bsum = bias1[nn];
            #pragma unroll
            for (int i = 0; i < 8; ++i) {
                float v = acc[i] + bsum;
                v = 0.5f * v * (1.0f + erff(v * 0.70710678118654752440f));
                hT[m0 + mAdd + i][nn] = (__bf16)v;
            }
        }
        __syncthreads();

        // Stage 2: out(16x128) += h(16x64) @ W2c(64x128)
        #pragma unroll
        for (int ot = 0; ot < 8; ++ot) {
            #pragma unroll
            for (int kk = 0; kk < HT; kk += 32) {
                v16bf a  = load_frag(&hT[0][0],  m0,      kk, HT + 8);
                v16bf bb = load_frag(&w2T[0][0], ot * 16, kk, HT + 8);
                oacc[ot] = wmma_bf16(a, bb, oacc[ot]);
            }
        }
        __syncthreads();   // all waves done reading w1T/w2T/hT

        if (hc + HT < Hc) {
            commit();      // convert + store next chunk into the LDS tiles
            __syncthreads();
        }
    }

    // Epilogue: (acc + bias2) * gate, scatter-add to out[b, s, :]
    #pragma unroll
    for (int ot = 0; ot < 8; ++ot) {
        int n = ot * 16 + nL;
        float b2 = bias2[n];
        #pragma unroll
        for (int i = 0; i < 8; ++i) {
            int m = m0 + mAdd + i;
            float v = (oacc[ot][i] + b2) * rowGate[m];
            atomicAdd(&out[(((size_t)rowB[m] * Sc + rowTok[m]) * DOc) + n], v);
        }
    }
}

// ---------------------------------------------------------------------------
// Launcher
// ---------------------------------------------------------------------------
extern "C" void kernel_launch(void* const* d_in, const int* in_sizes, int n_in,
                              void* d_out, int out_size, void* d_ws, size_t ws_size,
                              hipStream_t stream) {
    const float* x  = (const float*)d_in[0];   // [B,S,D]
    const float* gw = (const float*)d_in[1];   // [D,E]
    const float* gb = (const float*)d_in[2];   // [E]
    const float* w1 = (const float*)d_in[3];   // [E,D+1,H]
    const float* w2 = (const float*)d_in[4];   // [E,H+1,DO]
    float* out = (float*)d_out;                // [B,S,DO]

    // Workspace: logits/probs (transposed) + topv + topi  (~8.25 MB)
    float* logitsT = (float*)d_ws;                              // [B,E,S]
    float* topv    = logitsT + (size_t)Bc * Ec * Sc;            // [B,E,K]
    int*   topi    = (int*)(topv + (size_t)Bc * Ec * Kc);       // [B,E,K]

    k_zero   <<<1024, 256, 0, stream>>>(out, Bc * Sc * DOc);
    k_gate   <<<Bc * Sc / 64, 256, 0, stream>>>(x, gw, gb, logitsT);
    k_softmax<<<Bc * Ec, 256, 0, stream>>>(logitsT);
    k_topk   <<<Bc * Ec, 256, 0, stream>>>(logitsT, topv, topi);
    k_expert <<<Ec * 4, 256, 0, stream>>>(x, w1, w2, topv, topi, out);
}
